// OldRouterModel_49675591746331
// MI455X (gfx1250) — matrix operand order repaired and checked by reading, not compiled, not run on hardware
//
#include <hip/hip_runtime.h>
#include <hip/hip_bf16.h>
#include <math.h>

// ---------------- problem constants ----------------
#define NN   2048
#define EE   8192
#define DD   128      // DN == DE == 128
#define HH   4
#define DH   32       // DD / HH
#define CC   10

typedef __bf16 bf16_t;
typedef __attribute__((ext_vector_type(16))) __bf16 v16bf;
typedef __attribute__((ext_vector_type(8)))  __bf16 v8bf;
typedef __attribute__((ext_vector_type(8)))  float  v8f;

union V16BF { v16bf v; v8bf h[2]; __bf16 e[16]; };
union V8F   { v8f   v; float  e[8];  };

#define WMMA_BF16(A_, B_, C_) \
  __builtin_amdgcn_wmma_f32_16x16x32_bf16(false, (A_), false, (B_), (short)0, (C_), false, false)

__device__ __forceinline__ float gelu_tanh(float x) {
  const float k0 = 0.7978845608028654f;   // sqrt(2/pi)
  return 0.5f * x * (1.0f + tanhf(k0 * (x + 0.044715f * x * x * x)));
}

// ---------------- scores: one wave per row, 128-dot ----------------
__global__ void row_dot_kernel(const float* __restrict__ X, const float* __restrict__ w,
                               float* __restrict__ out, int rows) {
  int wid  = (blockIdx.x * blockDim.x + threadIdx.x) >> 5;
  int lane = threadIdx.x & 31;
  if (wid >= rows) return;
  float acc = 0.f;
  #pragma unroll
  for (int d = lane; d < DD; d += 32) acc += X[wid * DD + d] * w[d];
  #pragma unroll
  for (int m = 16; m >= 1; m >>= 1) acc += __shfl_xor(acc, m, 32);
  if (lane == 0) out[wid] = acc;
}

// ---------------- k-th largest via bitonic sort (single WG, dyn LDS) ----------------
__global__ void kth_largest_kernel(const float* __restrict__ vals, int n, int k,
                                   float* __restrict__ out) {
  extern __shared__ float s[];
  for (int i = threadIdx.x; i < n; i += blockDim.x) s[i] = vals[i];
  __syncthreads();
  for (int sz = 2; sz <= n; sz <<= 1) {
    for (int j = sz >> 1; j > 0; j >>= 1) {
      for (int i = threadIdx.x; i < n; i += blockDim.x) {
        int ixj = i ^ j;
        if (ixj > i) {
          float a = s[i], b = s[ixj];
          bool descRegion = ((i & sz) == 0);
          if ((descRegion && a < b) || (!descRegion && a > b)) { s[i] = b; s[ixj] = a; }
        }
      }
      __syncthreads();
    }
  }
  if (threadIdx.x == 0) out[0] = s[k - 1];
}

// ---------------- edge mask ----------------
__global__ void mask_kernel(const float* __restrict__ nscore, const float* __restrict__ escore,
                            const int* __restrict__ src, const int* __restrict__ dst,
                            const float* __restrict__ nth, const float* __restrict__ eth,
                            float* __restrict__ maskf) {
  int e = blockIdx.x * blockDim.x + threadIdx.x;
  if (e >= EE) return;
  bool m = (escore[e] >= eth[0]) && (nscore[src[e]] >= nth[0]) && (nscore[dst[e]] >= nth[0]);
  maskf[e] = m ? 1.f : 0.f;
}

// ---------------- gather endpoint context (f32 -> bf16, row-major A operand) --------
__global__ void gather_kernel(const float* __restrict__ nf, const int* __restrict__ src,
                              const int* __restrict__ dst, bf16_t* __restrict__ g) {
  int e = blockIdx.x, d = threadIdx.x;   // block = 128 threads
  g[e * DD + d] = (bf16_t)(nf[src[e] * DD + d] + nf[dst[e] * DD + d]);
}

// ---------------- f32 weight -> WMMA-B fragment-contiguous packed bf16 --------------
// packed layout: Wp[((k>>4)*DD + c)*16 + (k&15)]  -> one 32B load per B fragment
__global__ void pack_b_kernel(const float* __restrict__ in, bf16_t* __restrict__ out) {
  int i = blockIdx.x * blockDim.x + threadIdx.x;
  if (i >= DD * DD) return;
  int k = i / DD, c = i % DD;
  out[(((k >> 4) * DD + c) << 4) + (k & 15)] = (bf16_t)in[i];
}

// ---------------- 16x128 row-block WMMA GEMM, K=128, templated epilogue -------------
// A: row-major [M,128] bf16; Bp: packed B layout (see pack_b_kernel)
// MODE 0: out = bf16(acc), row-major
// MODE 1: out = bf16(acc + add[r,c] * mask[r]), row-major      (e_in = ctx + ef*mask)
// MODE 2: out = bf16(acc), packed-B layout                     (V for attention)
// MODE 3: out = bf16(gelu(acc + bias[c])), row-major
template<int MODE>
__global__ __launch_bounds__(32, 1)
void gemm16_kernel(const bf16_t* __restrict__ A, const bf16_t* __restrict__ Bp,
                   bf16_t* __restrict__ out, const float* __restrict__ add,
                   const float* __restrict__ maskf, const float* __restrict__ bias) {
  const int lane = threadIdx.x & 31;
  const int half = lane >> 4;
  const int nc   = lane & 15;
  const int row0 = blockIdx.x * 16;
  const int arow = row0 + nc;

  V8F acc[8];
  #pragma unroll
  for (int cb = 0; cb < 8; ++cb)
    #pragma unroll
    for (int j = 0; j < 8; ++j) acc[cb].e[j] = 0.f;

  #pragma unroll
  for (int kb = 0; kb < 4; ++kb) {
    V16BF a;
    a.h[0] = *(const v8bf*)(A + arow * DD + kb * 32 + half * 16);
    a.h[1] = *(const v8bf*)(A + arow * DD + kb * 32 + half * 16 + 8);
    #pragma unroll
    for (int cb = 0; cb < 8; ++cb) {
      V16BF b;
      b.v = *(const v16bf*)(Bp + (((kb * 2 + half) * DD + cb * 16 + nc) << 4));
      acc[cb].v = WMMA_BF16(a.v, b.v, acc[cb].v);
    }
  }

  #pragma unroll
  for (int cb = 0; cb < 8; ++cb) {
    #pragma unroll
    for (int j = 0; j < 8; ++j) {
      int r = row0 + half * 8 + j;
      int c = cb * 16 + nc;
      float v = acc[cb].e[j];
      if (MODE == 1) v += add[r * DD + c] * maskf[r];
      if (MODE == 3) v = gelu_tanh(v + bias[c]);
      if (MODE == 2) out[(((r >> 4) * DD + c) << 4) + (r & 15)] = (bf16_t)v;
      else           out[r * DD + c] = (bf16_t)v;
    }
  }
}

// ---------------- flash edge-attention: 1 wave / (head, 16-query block) -------------
// q,k: row-major bf16 [E,128]; vp: packed-B layout bf16
__global__ __launch_bounds__(32, 1)
void edge_attn_kernel(const bf16_t* __restrict__ qb, const bf16_t* __restrict__ kbm,
                      const bf16_t* __restrict__ vp, const int* __restrict__ src,
                      const int* __restrict__ dst, bf16_t* __restrict__ out) {
  __shared__ bf16_t pl[16 * 64];          // P tile staging (C-layout -> A-layout)

  const int lane = threadIdx.x;           // blockDim = 32
  const int half = lane >> 4;
  const int nc   = lane & 15;
  const int h    = blockIdx.y;
  const int e0   = blockIdx.x * 16;
  const int hoff = h * DH;
  const float scale = 0.17677669529663687f;   // 1/sqrt(32)

  // loop-invariant query-side endpoints for this lane's 8 rows (broadcast loads)
  int sqr[8], dqr[8];
  #pragma unroll
  for (int j = 0; j < 8; ++j) {
    sqr[j] = src[e0 + half * 8 + j];
    dqr[j] = dst[e0 + half * 8 + j];
  }

  V16BF aq;
  aq.h[0] = *(const v8bf*)(qb + (e0 + nc) * DD + hoff + half * 16);
  aq.h[1] = *(const v8bf*)(qb + (e0 + nc) * DD + hoff + half * 16 + 8);

  float rm[8], rl[8];
  V8F o0, o1;
  #pragma unroll
  for (int j = 0; j < 8; ++j) { rm[j] = -1e30f; rl[j] = 0.f; o0.e[j] = 0.f; o1.e[j] = 0.f; }

  for (int f0 = 0; f0 < EE; f0 += 64) {
    // key-side endpoints for this lane's 4 key columns (cache-resident)
    int ski[4], dki[4];
    #pragma unroll
    for (int t = 0; t < 4; ++t) {
      ski[t] = src[f0 + t * 16 + nc];
      dki[t] = dst[f0 + t * 16 + nc];
    }

    // prefetch next key/value block while we compute this one
    if (f0 + 64 < EE) {
      __builtin_prefetch(kbm + (f0 + 64 + lane) * DD + hoff, 0, 3);
      __builtin_prefetch(vp + (((size_t)((f0 + 64) >> 4) * DD) << 4) + lane * 64, 0, 3);
    }

    // S = Q * K^T : four 16-key tiles, one WMMA each (K fragment is one 32B load)
    V8F s[4];
    #pragma unroll
    for (int t = 0; t < 4; ++t) {
      V16BF bk;
      bk.v = *(const v16bf*)(kbm + (f0 + t * 16 + nc) * DD + hoff + half * 16);
      v8f z = {};
      s[t].v = WMMA_BF16(aq.v, bk.v, z);
    }

    __syncthreads();                      // previous-iter P loads done before rewrite

    // mask + online softmax over the 64-key block
    #pragma unroll
    for (int j = 0; j < 8; ++j) {
      int m  = half * 8 + j;
      int ge = e0 + m;
      float x[4];
      #pragma unroll
      for (int t = 0; t < 4; ++t) {
        int fg = f0 + t * 16 + nc;
        bool a = (sqr[j] == ski[t]) | (sqr[j] == dki[t]) |
                 (dqr[j] == ski[t]) | (dqr[j] == dki[t]) | (ge == fg);
        x[t] = a ? s[t].e[j] * scale : -1e9f;
      }
      float c = fmaxf(fmaxf(x[0], x[1]), fmaxf(x[2], x[3]));
      #pragma unroll
      for (int mv = 8; mv >= 1; mv >>= 1) c = fmaxf(c, __shfl_xor(c, mv, 32));
      float nm = fmaxf(rm[j], c);
      float al = __expf(rm[j] - nm);
      rm[j] = nm;

      float ps = 0.f;
      #pragma unroll
      for (int t = 0; t < 4; ++t) {
        float p = __expf(x[t] - nm);
        ps += p;
        pl[m * 64 + t * 16 + nc] = (bf16_t)p;
      }
      #pragma unroll
      for (int mv = 8; mv >= 1; mv >>= 1) ps += __shfl_xor(ps, mv, 32);
      rl[j] = rl[j] * al + ps;
      o0.e[j] *= al;
      o1.e[j] *= al;
    }
    __syncthreads();                      // P stores visible before re-fragmentation

    // O += P * V : two K=32 sub-blocks; P from LDS (2x ds_load_b128),
    // V fragments are single 32B loads from the packed layout
    #pragma unroll
    for (int sub = 0; sub < 2; ++sub) {
      V16BF ap;
      const bf16_t* pb = pl + nc * 64 + sub * 32 + half * 16;
      ap.h[0] = *(const v8bf*)(pb);
      ap.h[1] = *(const v8bf*)(pb + 8);
      int fb = (f0 >> 4) + sub * 2 + half;   // 16-key block index into packed V
      V16BF bv0, bv1;
      bv0.v = *(const v16bf*)(vp + (((size_t)fb * DD + hoff + nc) << 4));
      bv1.v = *(const v16bf*)(vp + (((size_t)fb * DD + hoff + 16 + nc) << 4));
      o0.v = WMMA_BF16(ap.v, bv0.v, o0.v);
      o1.v = WMMA_BF16(ap.v, bv1.v, o1.v);
    }
  }

  // finalize
  #pragma unroll
  for (int j = 0; j < 8; ++j) {
    float inv = 1.f / rl[j];
    int r = e0 + half * 8 + j;
    out[r * DD + hoff + nc]      = (bf16_t)(o0.e[j] * inv);
    out[r * DD + hoff + 16 + nc] = (bf16_t)(o1.e[j] * inv);
  }
}

// ---------------- classifier head: logits = h1 @ w2 + b2 ----------------
__global__ void head_kernel(const bf16_t* __restrict__ h1, const float* __restrict__ w2,
                            const float* __restrict__ b2, float* __restrict__ logits) {
  int idx = blockIdx.x * blockDim.x + threadIdx.x;
  if (idx >= EE * CC) return;
  int e = idx / CC, c = idx % CC;
  float acc = b2[c];
  #pragma unroll 4
  for (int d = 0; d < DD; ++d) acc += (float)h1[e * DD + d] * w2[d * CC + c];
  logits[idx] = acc;
}

// ---------------- host orchestration ----------------
extern "C" void kernel_launch(void* const* d_in, const int* in_sizes, int n_in,
                              void* d_out, int out_size, void* d_ws, size_t ws_size,
                              hipStream_t stream) {
  (void)in_sizes; (void)n_in; (void)out_size; (void)ws_size;
  const float* nf   = (const float*)d_in[0];
  const float* ef   = (const float*)d_in[1];
  const int*   eidx = (const int*)  d_in[2];
  const int*   srcp = eidx;
  const int*   dstp = eidx + EE;
  const float* wns  = (const float*)d_in[4];
  const float* wes  = (const float*)d_in[5];
  const float* wnp  = (const float*)d_in[6];
  const float* wq   = (const float*)d_in[7];
  const float* wk   = (const float*)d_in[8];
  const float* wv   = (const float*)d_in[9];
  const float* wo   = (const float*)d_in[10];
  const float* w1   = (const float*)d_in[11];
  const float* b1   = (const float*)d_in[12];
  const float* w2   = (const float*)d_in[13];
  const float* b2   = (const float*)d_in[14];
  float* logits = (float*)d_out;

  // workspace carve-up (256-byte aligned)
  char* cur = (char*)d_ws;
  auto carve = [&](size_t bytes) -> void* {
    void* p = cur;
    cur += (bytes + 255) & ~size_t(255);
    return p;
  };
  float*  nscore  = (float*) carve(NN * 4);
  float*  escore  = (float*) carve(EE * 4);
  float*  nth     = (float*) carve(256);
  float*  eth     = (float*) carve(256);
  float*  maskf   = (float*) carve(EE * 4);
  bf16_t* g_bf    = (bf16_t*)carve((size_t)EE * DD * 2);
  bf16_t* wnp_p   = (bf16_t*)carve(DD * DD * 2);
  bf16_t* wq_p    = (bf16_t*)carve(DD * DD * 2);
  bf16_t* wk_p    = (bf16_t*)carve(DD * DD * 2);
  bf16_t* wv_p    = (bf16_t*)carve(DD * DD * 2);
  bf16_t* wo_p    = (bf16_t*)carve(DD * DD * 2);
  bf16_t* w1_p    = (bf16_t*)carve(DD * DD * 2);
  bf16_t* ein_bf  = (bf16_t*)carve((size_t)EE * DD * 2);
  bf16_t* q_bf    = (bf16_t*)carve((size_t)EE * DD * 2);
  bf16_t* k_bf    = (bf16_t*)carve((size_t)EE * DD * 2);
  bf16_t* v_pk    = (bf16_t*)carve((size_t)EE * DD * 2);   // packed-B layout
  bf16_t* attn_bf = (bf16_t*)carve((size_t)EE * DD * 2);
  bf16_t* x2_bf   = (bf16_t*)carve((size_t)EE * DD * 2);
  bf16_t* h1_bf   = (bf16_t*)carve((size_t)EE * DD * 2);

  // 1) scores
  row_dot_kernel<<<NN / 8, 256, 0, stream>>>(nf, wns, nscore, NN);
  row_dot_kernel<<<EE / 8, 256, 0, stream>>>(ef, wes, escore, EE);

  // 2) exact top-k thresholds (descending bitonic sort in LDS)
  kth_largest_kernel<<<1, 1024, NN * sizeof(float), stream>>>(nscore, NN, NN / 2, nth);
  kth_largest_kernel<<<1, 1024, EE * sizeof(float), stream>>>(escore, EE, EE / 2, eth);

  // 3) edge mask + endpoint gather + weight packing (B-fragment contiguous)
  mask_kernel<<<(EE + 255) / 256, 256, 0, stream>>>(nscore, escore, srcp, dstp, nth, eth, maskf);
  gather_kernel<<<EE, DD, 0, stream>>>(nf, srcp, dstp, g_bf);
  pack_b_kernel<<<(DD * DD + 255) / 256, 256, 0, stream>>>(wnp, wnp_p);
  pack_b_kernel<<<(DD * DD + 255) / 256, 256, 0, stream>>>(wq,  wq_p);
  pack_b_kernel<<<(DD * DD + 255) / 256, 256, 0, stream>>>(wk,  wk_p);
  pack_b_kernel<<<(DD * DD + 255) / 256, 256, 0, stream>>>(wv,  wv_p);
  pack_b_kernel<<<(DD * DD + 255) / 256, 256, 0, stream>>>(wo,  wo_p);
  pack_b_kernel<<<(DD * DD + 255) / 256, 256, 0, stream>>>(w1,  w1_p);

  const int gemmGrid = EE / 16;   // 16x128 row block per wave
  // 4) e_in = (nf[src]+nf[dst]) @ wnp + ef * mask
  gemm16_kernel<1><<<gemmGrid, 32, 0, stream>>>(g_bf, wnp_p, ein_bf, ef, maskf, nullptr);
  // 5) q, k (row-major), v (packed for P@V B operand)
  gemm16_kernel<0><<<gemmGrid, 32, 0, stream>>>(ein_bf, wq_p, q_bf, nullptr, nullptr, nullptr);
  gemm16_kernel<0><<<gemmGrid, 32, 0, stream>>>(ein_bf, wk_p, k_bf, nullptr, nullptr, nullptr);
  gemm16_kernel<2><<<gemmGrid, 32, 0, stream>>>(ein_bf, wv_p, v_pk, nullptr, nullptr, nullptr);

  // 6) flash attention over edge-adjacency graph
  dim3 attnGrid(EE / 16, HH);
  edge_attn_kernel<<<attnGrid, 32, 0, stream>>>(q_bf, k_bf, v_pk, srcp, dstp, attn_bf);

  // 7) output proj + MLP
  gemm16_kernel<0><<<gemmGrid, 32, 0, stream>>>(attn_bf, wo_p, x2_bf, nullptr, nullptr, nullptr);
  gemm16_kernel<3><<<gemmGrid, 32, 0, stream>>>(x2_bf, w1_p, h1_bf, nullptr, nullptr, b1);

  // 8) classifier head
  head_kernel<<<(EE * CC + 255) / 256, 256, 0, stream>>>(h1_bf, w2, b2, logits);
}
